// SelfAttention_1185410974397
// MI455X (gfx1250) — matrix operand order repaired and checked by reading, not compiled, not run on hardware
//
#include <hip/hip_runtime.h>

typedef __attribute__((ext_vector_type(16))) __bf16 bf16x16;
typedef __attribute__((ext_vector_type(8)))  __bf16 bf16x8;
typedef __attribute__((ext_vector_type(4)))  __bf16 bf16x4;
typedef __attribute__((ext_vector_type(8)))  float  f32x8;
typedef __attribute__((ext_vector_type(4)))  unsigned int u32x4;
typedef __attribute__((ext_vector_type(8)))  int i32x8;
typedef __attribute__((ext_vector_type(4)))  int i32x4;

#define S_LEN  2048
#define DMODEL 1024
#define NHEAD  16
#define DHEAD  64
#define NROW   4096   /* B * S */

#if defined(__has_builtin)
#if __has_builtin(__builtin_amdgcn_tensor_load_to_lds)
#define USE_TDM 1
#endif
#endif

// ---- WMMA fragment helpers (layouts per CDNA5 ISA) -------------
// A 16x32 bf16: lane m=lane&15; VGPR v holds K pair (v&4)*4 + half*8 + (v&3)*2
__device__ __forceinline__ int kbaseA(int v, int half) {
  return ((v & 4) << 2) + half * 8 + (v & 3) * 2;
}

__device__ __forceinline__ f32x8 wmma_bf16(bf16x16 a, bf16x16 b, f32x8 c) {
  return __builtin_amdgcn_wmma_f32_16x16x32_bf16(false, a, false, b, (short)0, c,
                                                 false, false);
}

__device__ __forceinline__ bf16x16 join8(bf16x8 lo, bf16x8 hi) {
  bf16x16 r;
#pragma unroll
  for (int i = 0; i < 8; ++i) { r[i] = lo[i]; r[8 + i] = hi[i]; }
  return r;
}

// A-frag from bf16 row: VGPR0-3 = k[half*8 .. +7], VGPR4-7 = k[16+half*8 .. +7]
__device__ __forceinline__ bf16x16 frag_a(const __bf16* p, int half) {
  return join8(*(const bf16x8*)(p + half * 8),
               *(const bf16x8*)(p + 16 + half * 8));
}
// B-frag where this lane's column is contiguous in K: VGPR0-3 = k[half*8 .. +7], VGPR4-7 = k[16+half*8 .. +7]
__device__ __forceinline__ bf16x16 frag_b(const __bf16* p, int half) {
  return join8(*(const bf16x8*)(p + half * 8),
               *(const bf16x8*)(p + 16 + half * 8));
}

// ---- Tensor Data Mover: 2D tile -> LDS (D# per CDNA5 ISA) -----
__device__ __forceinline__ unsigned lds_off(const void* p) {
  return (unsigned)(unsigned long long)(uintptr_t)p;  // LDS aperture: addr[31:0]
}

#ifdef USE_TDM
__device__ __forceinline__ void tdm_load_2d(unsigned ldsa, const __bf16* gaddr,
                                            unsigned tens_d0, unsigned tens_d1,
                                            unsigned tile_d0, unsigned tile_d1,
                                            unsigned stride0 /*elements*/) {
  const unsigned long long ga = (unsigned long long)(uintptr_t)gaddr;
  u32x4 g0;
  g0[0] = 1u;                                             // count=1
  g0[1] = ldsa;                                           // lds_addr (bytes)
  g0[2] = (unsigned)(ga & 0xFFFFFFFFu);                   // global_addr[31:0]
  g0[3] = (unsigned)((ga >> 32) & 0x01FFFFFFu) | (2u << 30);  // addr[56:32]|type=2
  i32x8 g1;
  g1[0] = (int)(1u << 16);                                // data_size=1 (2B), mask=0
  g1[1] = (int)((tens_d0 & 0xFFFFu) << 16);               // tensor_dim0[15:0]
  g1[2] = (int)((tens_d0 >> 16) | ((tens_d1 & 0xFFFFu) << 16));
  g1[3] = (int)((tens_d1 >> 16) | (tile_d0 << 16));       // tile_dim0
  g1[4] = (int)(tile_d1 & 0xFFFFu);                       // tile_dim1 (tile_dim2=0)
  g1[5] = (int)stride0;                                   // dim0 stride[31:0]
  g1[6] = 0;
  g1[7] = 0;
  const i32x4 z4 = {0, 0, 0, 0};                          // 2D: groups 2/3 unused
  const i32x8 z8 = {0, 0, 0, 0, 0, 0, 0, 0};
  __builtin_amdgcn_tensor_load_to_lds(g0, g1, z4, z4, z8, 0);
}
#endif

// ---- Kernel: f32 -> bf16 streaming convert --------------------------------
__global__ __launch_bounds__(256)
void cvt_bf16_kernel(const float* __restrict__ in, __bf16* __restrict__ out) {
  const int i = blockIdx.x * 256 + threadIdx.x;
  const float4 f = ((const float4*)in)[i];
  bf16x4 o;
  o[0] = (__bf16)f.x; o[1] = (__bf16)f.y; o[2] = (__bf16)f.z; o[3] = (__bf16)f.w;
  ((bf16x4*)out)[i] = o;
}

// ---- Kernel: W[k][n] f32 -> Wt[n][k] bf16 (32x32 LDS tile transpose) ------
__global__ __launch_bounds__(256)
void transpose_cvt_kernel(const float* __restrict__ W, __bf16* __restrict__ Wt) {
  __shared__ float tile[32][33];
  const int tx = threadIdx.x & 31, ty = threadIdx.x >> 5;
  const int c0 = blockIdx.x * 32, r0 = blockIdx.y * 32;
#pragma unroll
  for (int i = ty; i < 32; i += 8)
    tile[i][tx] = W[(size_t)(r0 + i) * DMODEL + c0 + tx];
  __syncthreads();
#pragma unroll
  for (int i = ty; i < 32; i += 8)
    Wt[(size_t)(c0 + i) * DMODEL + r0 + tx] = (__bf16)tile[tx][i];
}

// ---- Kernel: O = xb @ Wt^T + bias (bf16 in, bf16 out) ---------------------
// Block = 8 waves, tile 32x64; each wave one 16x16 tile, K-loop of 32.
// vtrans=0: O[row][col] row-major.  vtrans=1: O[col][token] (V transposed).
__global__ __launch_bounds__(256)
void qkv_gemm_kernel(const __bf16* __restrict__ xb, const __bf16* __restrict__ Wt,
                     const float* __restrict__ bias, __bf16* __restrict__ O,
                     int vtrans) {
  const int lane = threadIdx.x & 31;
  const int wv   = threadIdx.x >> 5;
  const int n    = lane & 15;
  const int half = lane >> 4;
  const int row0 = blockIdx.y * 32 + (wv & 1) * 16;
  const int col0 = blockIdx.x * 64 + (wv >> 1) * 16;

  f32x8 acc = {};
  const __bf16* arow = xb + (size_t)(row0 + n) * DMODEL;
  const __bf16* brow = Wt + (size_t)(col0 + n) * DMODEL;

  for (int kk = 0; kk < DMODEL; kk += 32)
    acc = wmma_bf16(frag_a(arow + kk, half), frag_b(brow + kk, half), acc);

  const float bn = bias[col0 + n];
  if (vtrans == 0) {
#pragma unroll
    for (int v = 0; v < 8; ++v)
      O[(size_t)(row0 + v + 8 * half) * DMODEL + col0 + n] = (__bf16)(acc[v] + bn);
  } else {
    bf16x8 t;
#pragma unroll
    for (int v = 0; v < 8; ++v) t[v] = (__bf16)(acc[v] + bn);
    *(bf16x8*)(O + (size_t)(col0 + n) * NROW + row0 + 8 * half) = t;
  }
}

// ---- Kernel: streaming-softmax attention ----------------------------------
// Block = 8 waves sharing (b,h); wave wv owns q-tile (qg*8+wv). K/V 32-key
// chunks DMA'd to LDS by the TDM, double-buffered, overlapped with compute.
__global__ __launch_bounds__(256)
void flash_attn_kernel(const __bf16* __restrict__ Q, const __bf16* __restrict__ K,
                       const __bf16* __restrict__ Vt, float* __restrict__ O) {
  __shared__ __align__(16) __bf16 kt[2][32][64];   // [key][d]
  __shared__ __align__(16) __bf16 vt[2][64][32];   // [d][key]
  __shared__ __align__(16) __bf16 pslab[8][16][32];

  const int lane = threadIdx.x & 31;
  const int wv   = threadIdx.x >> 5;
  const int n    = lane & 15;
  const int half = lane >> 4;
  const int qg   = blockIdx.x & 15;
  const int h    = (blockIdx.x >> 4) & (NHEAD - 1);
  const int b    = blockIdx.x >> 8;

  const int qrow0 = b * S_LEN + (qg * 8 + wv) * 16;
  const __bf16* Qb = Q + (size_t)qrow0 * DMODEL + h * DHEAD;
  const __bf16* Kb = K + (size_t)(b * S_LEN) * DMODEL + h * DHEAD;
  const __bf16* Vb = Vt + (size_t)(h * DHEAD) * NROW + b * S_LEN;

  // Q fragments (16 q-rows x 64 d) resident in registers
  const bf16x16 aq0 = frag_a(Qb + (size_t)n * DMODEL, half);
  const bf16x16 aq1 = frag_a(Qb + (size_t)n * DMODEL + 32, half);

  f32x8 acc0 = {}, acc1 = {}, acc2 = {}, acc3 = {};
  float mrow[8], lrow[8];
#pragma unroll
  for (int v = 0; v < 8; ++v) { mrow[v] = -3.0e38f; lrow[v] = 0.0f; }
  const float scale = 0.125f;   // 1/sqrt(64)

  const int NCHUNK = S_LEN / 32;

#ifdef USE_TDM
  if (wv == 0) {  // prologue: DMA tile 0 into buffer 0
    tdm_load_2d(lds_off(&kt[0][0][0]), Kb, DMODEL, NROW, 64, 32, DMODEL);
    tdm_load_2d(lds_off(&vt[0][0][0]), Vb, NROW, DMODEL, 32, 64, NROW);
  }
#endif

  for (int i = 0; i < NCHUNK; ++i) {
    const int cur = i & 1;
#ifdef USE_TDM
    if (wv == 0) {
      if (i + 1 < NCHUNK) {
        const int key1 = (i + 1) * 32, nxt = (i + 1) & 1;
        tdm_load_2d(lds_off(&kt[nxt][0][0]), Kb + (size_t)key1 * DMODEL,
                    DMODEL, NROW, 64, 32, DMODEL);
        tdm_load_2d(lds_off(&vt[nxt][0][0]), Vb + key1,
                    NROW, DMODEL, 32, 64, NROW);
        __builtin_amdgcn_s_wait_tensorcnt(2);   // tile i complete (in-order)
      } else {
        __builtin_amdgcn_s_wait_tensorcnt(0);
      }
    }
    __syncthreads();                            // tile `cur` visible to all
#else
    __syncthreads();
    {   // cooperative fallback copy: 256 threads, b128 each
      const int key0 = i * 32;
      const int t = threadIdx.x;
      {
        const int r = t >> 3, c8 = (t & 7) * 8;           // 32 x 64
        *(bf16x8*)(&kt[cur][r][c8]) =
            *(const bf16x8*)(Kb + (size_t)(key0 + r) * DMODEL + c8);
      }
      {
        const int r = t >> 2, c8 = (t & 3) * 8;           // 64 x 32
        *(bf16x8*)(&vt[cur][r][c8]) =
            *(const bf16x8*)(Vb + (size_t)r * NROW + key0 + c8);
      }
    }
    __syncthreads();
#endif

    // scores: 16 q-rows x 32 keys as two 16x16 tiles, K-dim = d (2 steps)
    f32x8 c0 = {}, c1 = {};
    c0 = wmma_bf16(aq0, frag_b(&kt[cur][n][0],       half), c0);
    c0 = wmma_bf16(aq1, frag_b(&kt[cur][n][32],      half), c0);
    c1 = wmma_bf16(aq0, frag_b(&kt[cur][n + 16][0],  half), c1);
    c1 = wmma_bf16(aq1, frag_b(&kt[cur][n + 16][32], half), c1);

    float p0[8], p1[8], alpha[8];
#pragma unroll
    for (int v = 0; v < 8; ++v) {
      const float s0 = c0[v] * scale;
      const float s1 = c1[v] * scale;
      float cm = fmaxf(s0, s1);                 // row max over 16 lanes of half
      cm = fmaxf(cm, __shfl_xor(cm, 8));
      cm = fmaxf(cm, __shfl_xor(cm, 4));
      cm = fmaxf(cm, __shfl_xor(cm, 2));
      cm = fmaxf(cm, __shfl_xor(cm, 1));
      const float mnew = fmaxf(mrow[v], cm);
      alpha[v] = __expf(mrow[v] - mnew);
      p0[v] = __expf(s0 - mnew);
      p1[v] = __expf(s1 - mnew);
      float rs = p0[v] + p1[v];
      rs += __shfl_xor(rs, 8);
      rs += __shfl_xor(rs, 4);
      rs += __shfl_xor(rs, 2);
      rs += __shfl_xor(rs, 1);
      lrow[v] = lrow[v] * alpha[v] + rs;
      mrow[v] = mnew;
    }
#pragma unroll
    for (int v = 0; v < 8; ++v) {
      acc0[v] *= alpha[v]; acc1[v] *= alpha[v];
      acc2[v] *= alpha[v]; acc3[v] *= alpha[v];
    }

    // transpose P (C-layout -> A-layout) through this wave's private slab
#pragma unroll
    for (int v = 0; v < 8; ++v) {
      const int r = v + 8 * half;
      pslab[wv][r][n]      = (__bf16)p0[v];
      pslab[wv][r][n + 16] = (__bf16)p1[v];
    }
    asm volatile("s_wait_dscnt 0" ::: "memory");  // wave-private: DS in-order
    bf16x16 pf;
#pragma unroll
    for (int v = 0; v < 8; ++v) {
      const int k = kbaseA(v, half);
      pf[2 * v]     = pslab[wv][n][k];
      pf[2 * v + 1] = pslab[wv][n][k + 1];
    }

    // out += P @ V   (B[k=key][n=d], keys contiguous in vt rows)
    acc0 = wmma_bf16(pf, frag_b(&vt[cur][n][0],      half), acc0);
    acc1 = wmma_bf16(pf, frag_b(&vt[cur][16 + n][0], half), acc1);
    acc2 = wmma_bf16(pf, frag_b(&vt[cur][32 + n][0], half), acc2);
    acc3 = wmma_bf16(pf, frag_b(&vt[cur][48 + n][0], half), acc3);

    __syncthreads();                            // done with tile `cur`
  }

  // normalize and store fp32 [B,S,H*Dh]
  float* Ob = O + (size_t)qrow0 * DMODEL + h * DHEAD;
#pragma unroll
  for (int v = 0; v < 8; ++v) {
    const float inv = 1.0f / lrow[v];
    const size_t r = (size_t)(v + 8 * half) * DMODEL;
    Ob[r + n]      = acc0[v] * inv;
    Ob[r + 16 + n] = acc1[v] * inv;
    Ob[r + 32 + n] = acc2[v] * inv;
    Ob[r + 48 + n] = acc3[v] * inv;
  }
}

// ---- launch ---------------------------------------------------------------
extern "C" void kernel_launch(void* const* d_in, const int* in_sizes, int n_in,
                              void* d_out, int out_size, void* d_ws, size_t ws_size,
                              hipStream_t stream) {
  (void)in_sizes; (void)n_in; (void)out_size; (void)ws_size;
  const float* x  = (const float*)d_in[0];
  const float* Wq = (const float*)d_in[1];
  const float* bq = (const float*)d_in[2];
  const float* Wk = (const float*)d_in[3];
  const float* bk = (const float*)d_in[4];
  const float* Wv = (const float*)d_in[5];
  const float* bv = (const float*)d_in[6];

  char* w = (char*)d_ws;
  __bf16* xb  = (__bf16*)w;  w += (size_t)NROW * DMODEL * 2;     // 8 MB
  __bf16* Wtq = (__bf16*)w;  w += (size_t)DMODEL * DMODEL * 2;   // 2 MB
  __bf16* Wtk = (__bf16*)w;  w += (size_t)DMODEL * DMODEL * 2;
  __bf16* Wtv = (__bf16*)w;  w += (size_t)DMODEL * DMODEL * 2;
  __bf16* Qb  = (__bf16*)w;  w += (size_t)NROW * DMODEL * 2;     // 8 MB
  __bf16* Kb  = (__bf16*)w;  w += (size_t)NROW * DMODEL * 2;     // 8 MB
  __bf16* Vtb = (__bf16*)w;  w += (size_t)DMODEL * NROW * 2;     // 8 MB

  cvt_bf16_kernel<<<(NROW * DMODEL) / (4 * 256), 256, 0, stream>>>(x, xb);
  dim3 gt(DMODEL / 32, DMODEL / 32);
  transpose_cvt_kernel<<<gt, 256, 0, stream>>>(Wq, Wtq);
  transpose_cvt_kernel<<<gt, 256, 0, stream>>>(Wk, Wtk);
  transpose_cvt_kernel<<<gt, 256, 0, stream>>>(Wv, Wtv);

  dim3 gp(DMODEL / 64, NROW / 32);
  qkv_gemm_kernel<<<gp, 256, 0, stream>>>(xb, Wtq, bq, Qb, 0);
  qkv_gemm_kernel<<<gp, 256, 0, stream>>>(xb, Wtk, bk, Kb, 0);
  qkv_gemm_kernel<<<gp, 256, 0, stream>>>(xb, Wtv, bv, Vtb, 1);

  // blocks: b(2) x h(16) x qgroup(16) = 512, 8 waves each
  flash_attn_kernel<<<512, 256, 0, stream>>>(Qb, Kb, Vtb, (float*)d_out);
}
